// Block_47734266527937
// MI455X (gfx1250) — matrix operand order repaired and checked by reading, not compiled, not run on hardware
//
#include <hip/hip_runtime.h>
#include <hip/hip_bf16.h>

// Transformer block (pre-LN attention + FFN), B=4 T=2048 C=1024 H=16 DH=64.
// All GEMMs + attention use v_wmma_f32_16x16x32_f16 (wave32, gfx1250).
// Tile staging uses global_load_async_to_lds_b128 (ASYNCcnt) on gfx1250.

typedef __attribute__((ext_vector_type(16))) _Float16 v16h;
typedef __attribute__((ext_vector_type(8)))  _Float16 v8h;
typedef __attribute__((ext_vector_type(4)))  _Float16 v4h;
typedef __attribute__((ext_vector_type(8)))  float    v8f;

#define NB_T   2048
#define NB_C   1024
#define NB_H   16
#define NB_DH  64
#define NB_M   8192   // B*T

// Async copy: 16 bytes global -> LDS per active lane. The low 32 bits of a
// flat pointer into __shared__ are the workgroup-relative LDS byte offset
//, which is what VDST must hold.
#define ASYNC_B128(ldsp, gp)                                                 \
  asm volatile("global_load_async_to_lds_b128 %0, %1, off"                   \
               :: "v"((unsigned)(size_t)(ldsp)), "v"(gp) : "memory")
#define ASYNC_WAIT() asm volatile("s_wait_asynccnt 0" ::: "memory")

// ---------------------------------------------------------------- fragments
__device__ __forceinline__ v16h frag_from2(v8h lo, v8h hi) {
  v16h r;
#pragma unroll
  for (int i = 0; i < 8; ++i) { r[i] = lo[i]; r[i + 8] = hi[i]; }
  return r;
}

// A-fragment (16x32 f16), tile laid out [row][stride] row-major in K.
// Per ISA: lanes 0-15 row m=lane, halves 0-7 -> K=k0..k0+7, 8-15 -> K=k0+16..k0+23;
// lanes 16-31 same rows, K offset +8.
__device__ __forceinline__ v16h load_a_frag(const _Float16* s, int stride,
                                            int mBase, int k0, int lane) {
  const _Float16* p = s + (mBase + (lane & 15)) * stride + k0 + ((lane >> 4) << 3);
  return frag_from2(*(const v8h*)p, *(const v8h*)(p + 16));
}

// B-fragment (32x16 f16), tile laid out [col][stride] (K contiguous per column).
// lanes 0-15: n=lane, K=k0..k0+7, k0+16..k0+23; lanes 16-31: n=lane-16, K offset +8.
__device__ __forceinline__ v16h load_b_frag(const _Float16* s, int stride,
                                            int nBase, int k0, int lane) {
  const _Float16* p = s + (nBase + (lane & 15)) * stride + k0 + ((lane >> 4) << 3);
  return frag_from2(*(const v8h*)p, *(const v8h*)(p + 16));
}

// ------------------------------------------------- f32 -> f16 + transpose
// in: f32 [K][N] row-major;  out: f16 [N][K] row-major (i.e. W^T).
__global__ __launch_bounds__(256) void cvt_t_kernel(const float* __restrict__ in,
                                                    _Float16* __restrict__ out,
                                                    int K, int N) {
  __shared__ float tile[32][33];
  const int nb = blockIdx.x * 32, kb = blockIdx.y * 32;
  const int tx = threadIdx.x & 31, ty = threadIdx.x >> 5;   // 32 x 8
#pragma unroll
  for (int i = 0; i < 32; i += 8)
    tile[ty + i][tx] = in[(size_t)(kb + ty + i) * N + nb + tx];
  __syncthreads();
#pragma unroll
  for (int i = 0; i < 32; i += 8)
    out[(size_t)(nb + ty + i) * K + kb + tx] = (_Float16)tile[tx][ty + i];
}

// ---------------------------------------------------------------- LayerNorm
// One wave (32 lanes) per row of 1024; writes f16.
__global__ __launch_bounds__(256) void ln_kernel(const float* __restrict__ x,
                                                 const float* __restrict__ g,
                                                 const float* __restrict__ be,
                                                 _Float16* __restrict__ out) {
  const int row  = blockIdx.x * 8 + (threadIdx.x >> 5);
  const int lane = threadIdx.x & 31;
  const float4* xr = (const float4*)(x + (size_t)row * NB_C);
  float4 v[8];
  float s = 0.f;
#pragma unroll
  for (int i = 0; i < 8; ++i) {
    v[i] = xr[i * 32 + lane];
    s += v[i].x + v[i].y + v[i].z + v[i].w;
  }
#pragma unroll
  for (int m = 1; m < 32; m <<= 1) s += __shfl_xor(s, m);
  const float mu = s * (1.f / NB_C);
  float ss = 0.f;
#pragma unroll
  for (int i = 0; i < 8; ++i) {
    float dx;
    dx = v[i].x - mu; ss += dx * dx;
    dx = v[i].y - mu; ss += dx * dx;
    dx = v[i].z - mu; ss += dx * dx;
    dx = v[i].w - mu; ss += dx * dx;
  }
#pragma unroll
  for (int m = 1; m < 32; m <<= 1) ss += __shfl_xor(ss, m);
  const float rs = rsqrtf(ss * (1.f / NB_C) + 1e-5f);
  const float4* gr = (const float4*)g;
  const float4* br = (const float4*)be;
  v4h* orow = (v4h*)(out + (size_t)row * NB_C);
#pragma unroll
  for (int i = 0; i < 8; ++i) {
    const int i4 = i * 32 + lane;
    const float4 gg = gr[i4], bb = br[i4];
    v4h o;
    o[0] = (_Float16)((v[i].x - mu) * rs * gg.x + bb.x);
    o[1] = (_Float16)((v[i].y - mu) * rs * gg.y + bb.y);
    o[2] = (_Float16)((v[i].z - mu) * rs * gg.z + bb.z);
    o[3] = (_Float16)((v[i].w - mu) * rs * gg.w + bb.w);
    orow[i4] = o;
  }
}

// ---------------------------------------------------------------- WMMA GEMM
// D = A[M,K] @ W^T[N,K]^T. 128x128 block tile, K-step 64. 256 threads = 8
// waves in a 4x2 grid; each wave computes 32x64 (2x4 fragments of 16x16).
// Both LDS tiles are [row][GM_KP] with K contiguous -> async b128 copies.
#define GM_MB 128
#define GM_NB 128
#define GM_KB 64
#define GM_KP 72   // LDS row stride (halves) = 144B: 16B-aligned, conflict-free

enum { OUT_QK = 0, OUT_VT = 1, OUT_RESID = 2, OUT_RELU = 3 };

template <int MODE>
__global__ __launch_bounds__(256) void gemm_kernel(const _Float16* __restrict__ A,
                                                   const _Float16* __restrict__ WT,
                                                   const float* __restrict__ bias,
                                                   const float* __restrict__ resid,
                                                   void* __restrict__ outp,
                                                   int N, int K) {
  __shared__ _Float16 sA[GM_MB * GM_KP];
  __shared__ _Float16 sB[GM_NB * GM_KP];
  const int tid = threadIdx.x, lane = tid & 31, wv = tid >> 5;
  const int mBlock = blockIdx.y * GM_MB, nBlock = blockIdx.x * GM_NB;
  const int wRow = (wv & 3) * 32, wCol = (wv >> 2) * 64;

  v8f acc[2][4];
#pragma unroll
  for (int i = 0; i < 2; ++i)
#pragma unroll
    for (int j = 0; j < 4; ++j)
#pragma unroll
      for (int e = 0; e < 8; ++e) acc[i][j][e] = 0.f;

  const int cRow = tid >> 1, cK = (tid & 1) * 32;   // 128 rows x 64 K per tile

  for (int kt = 0; kt < K; kt += GM_KB) {
    __syncthreads();
    const _Float16* ag = A  + (size_t)(mBlock + cRow) * K + kt + cK;
    const _Float16* bg = WT + (size_t)(nBlock + cRow) * K + kt + cK;
#pragma unroll
    for (int i = 0; i < 4; ++i)
      ASYNC_B128(&sA[cRow * GM_KP + cK + i * 8], ag + i * 8);
#pragma unroll
    for (int i = 0; i < 4; ++i)
      ASYNC_B128(&sB[cRow * GM_KP + cK + i * 8], bg + i * 8);
    if (kt + GM_KB < K) {
      __builtin_prefetch(ag + GM_KB, 0, 0);
      __builtin_prefetch(bg + GM_KB, 0, 0);
    }
    ASYNC_WAIT();
    __syncthreads();
#pragma unroll
    for (int k0 = 0; k0 < GM_KB; k0 += 32) {
      v16h af[2], bf[4];
#pragma unroll
      for (int i = 0; i < 2; ++i) af[i] = load_a_frag(sA, GM_KP, wRow + i * 16, k0, lane);
#pragma unroll
      for (int j = 0; j < 4; ++j) bf[j] = load_b_frag(sB, GM_KP, wCol + j * 16, k0, lane);
#pragma unroll
      for (int i = 0; i < 2; ++i)
#pragma unroll
        for (int j = 0; j < 4; ++j)
          acc[i][j] = __builtin_amdgcn_wmma_f32_16x16x32_f16(
              false, af[i], false, bf[j], (short)0, acc[i][j], false, false);
    }
  }

  // Epilogue. D layout: VGPR r = row r (lanes 0-15) / row r+8 (lanes 16-31), col = lane&15.
  const int nLoc = lane & 15, mHalf = (lane >> 4) * 8;
#pragma unroll
  for (int i = 0; i < 2; ++i) {
#pragma unroll
    for (int j = 0; j < 4; ++j) {
      const int col  = nBlock + wCol + j * 16 + nLoc;
      const int row0 = mBlock + wRow + i * 16 + mHalf;
      v8f c = acc[i][j];
      if constexpr (MODE == OUT_QK) {           // f16 [B,H,T,DH]
        _Float16* out = (_Float16*)outp;
        const int hh = col >> 6, d = col & 63;
#pragma unroll
        for (int r = 0; r < 8; ++r) {
          const int row = row0 + r, b = row >> 11, t = row & (NB_T - 1);
          out[((size_t)(b * NB_H + hh) * NB_T + t) * NB_DH + d] = (_Float16)c[r];
        }
      } else if constexpr (MODE == OUT_VT) {    // f16 [B,H,DH,T] (contiguous in t)
        _Float16* out = (_Float16*)outp;
        const int hh = col >> 6, d = col & 63;
        const int b = row0 >> 11, t0 = row0 & (NB_T - 1);
        v8h o;
#pragma unroll
        for (int r = 0; r < 8; ++r) o[r] = (_Float16)c[r];
        *(v8h*)&out[((size_t)(b * NB_H + hh) * NB_DH + d) * NB_T + t0] = o;
      } else if constexpr (MODE == OUT_RESID) { // f32 row-major, +bias +residual
        float* out = (float*)outp;
        const float bv = bias[col];
#pragma unroll
        for (int r = 0; r < 8; ++r) {
          const size_t idx = (size_t)(row0 + r) * N + col;
          out[idx] = c[r] + bv + resid[idx];
        }
      } else {                                  // OUT_RELU: f16 row-major, relu(x+bias)
        _Float16* out = (_Float16*)outp;
        const float bv = bias[col];
#pragma unroll
        for (int r = 0; r < 8; ++r) {
          float v = c[r] + bv;
          out[(size_t)(row0 + r) * N + col] = (_Float16)(v > 0.f ? v : 0.f);
        }
      }
    }
  }
}

// ---------------------------------------------------------------- attention
// Flash attention, causal. Grid: (T/128, B*H). 8 waves; wave w owns 16 query
// rows. K tile [s][dh], V tile [dh][s] in LDS via async; scale 1/8 folded
// into the Q fragments.
#define AT_BR 128
#define AT_BC 64
#define AT_KP 72

__global__ __launch_bounds__(256) void attn_kernel(const _Float16* __restrict__ Q,
                                                   const _Float16* __restrict__ Kh,
                                                   const _Float16* __restrict__ Vt,
                                                   _Float16* __restrict__ O) {
  __shared__ _Float16 sK[AT_BC * AT_KP];
  __shared__ _Float16 sV[NB_DH * AT_KP];
  __shared__ _Float16 sP[AT_BR * AT_KP];
  const int tid = threadIdx.x, lane = tid & 31, wv = tid >> 5;
  const int bh = blockIdx.y, r0 = blockIdx.x * AT_BR;
  const int wRow = wv * 16;
  const _Float16* qh = Q  + (size_t)bh * NB_T * NB_DH;
  const _Float16* kh = Kh + (size_t)bh * NB_T * NB_DH;
  const _Float16* vh = Vt + (size_t)bh * NB_DH * NB_T;

  // Q fragments (kept in registers, pre-scaled by DH^-0.5 = 1/8, exact in f16)
  v16h qf[2];
  {
    const int m = r0 + wRow + (lane & 15);
#pragma unroll
    for (int kk = 0; kk < 2; ++kk) {
      const _Float16* p = qh + (size_t)m * NB_DH + kk * 32 + ((lane >> 4) << 3);
      v16h f = frag_from2(*(const v8h*)p, *(const v8h*)(p + 16));
#pragma unroll
      for (int e = 0; e < 16; ++e) f[e] = f[e] * (_Float16)0.125f;
      qf[kk] = f;
    }
  }

  v8f Oa[4];
#pragma unroll
  for (int j = 0; j < 4; ++j)
#pragma unroll
    for (int e = 0; e < 8; ++e) Oa[j][e] = 0.f;
  float rMax[8], rSum[8];
#pragma unroll
  for (int r = 0; r < 8; ++r) { rMax[r] = -1e30f; rSum[r] = 0.f; }

  const int jMax = (r0 + AT_BR - 1) / AT_BC;
  const int kRow = tid >> 2, kOff = (tid & 3) * 16;

  for (int j = 0; j <= jMax; ++j) {
    const int cb = j * AT_BC;
    __syncthreads();
    {
      const _Float16* gp = kh + (size_t)(cb + kRow) * NB_DH + kOff;
      ASYNC_B128(&sK[kRow * AT_KP + kOff],     gp);
      ASYNC_B128(&sK[kRow * AT_KP + kOff + 8], gp + 8);
      const _Float16* vp = vh + (size_t)kRow * NB_T + cb + kOff;
      ASYNC_B128(&sV[kRow * AT_KP + kOff],     vp);
      ASYNC_B128(&sV[kRow * AT_KP + kOff + 8], vp + 8);
      ASYNC_WAIT();
    }
    __syncthreads();

    // S = (Q/8) @ K^T   (B-frag: n = key s, K-dim = dh, contiguous in sK rows)
    v8f S[4];
#pragma unroll
    for (int nf = 0; nf < 4; ++nf) {
      v8f a;
#pragma unroll
      for (int e = 0; e < 8; ++e) a[e] = 0.f;
#pragma unroll
      for (int kk = 0; kk < 2; ++kk) {
        v16h bf = load_b_frag(sK, AT_KP, nf * 16, kk * 32, lane);
        a = __builtin_amdgcn_wmma_f32_16x16x32_f16(false, qf[kk], false, bf,
                                                   (short)0, a, false, false);
      }
      S[nf] = a;
    }

    // causal mask (only on tiles touching the diagonal for this wave's rows)
    if (cb + AT_BC - 1 > r0 + wRow) {
#pragma unroll
      for (int nf = 0; nf < 4; ++nf)
#pragma unroll
        for (int r = 0; r < 8; ++r) {
          const int colg = cb + nf * 16 + (lane & 15);
          const int rowg = r0 + wRow + ((lane >> 4) << 3) + r;
          if (colg > rowg) S[nf][r] = -1e30f;
        }
    }

    // online softmax: row r lives in VGPR r across the 16 lanes of a half-wave
#pragma unroll
    for (int r = 0; r < 8; ++r) {
      float mx = fmaxf(fmaxf(S[0][r], S[1][r]), fmaxf(S[2][r], S[3][r]));
      mx = fmaxf(mx, __shfl_xor(mx, 1));
      mx = fmaxf(mx, __shfl_xor(mx, 2));
      mx = fmaxf(mx, __shfl_xor(mx, 4));
      mx = fmaxf(mx, __shfl_xor(mx, 8));
      const float nm   = fmaxf(rMax[r], mx);
      const float corr = __expf(rMax[r] - nm);
      rMax[r] = nm;
      float sm = 0.f;
#pragma unroll
      for (int nf = 0; nf < 4; ++nf) { S[nf][r] = __expf(S[nf][r] - nm); sm += S[nf][r]; }
      sm += __shfl_xor(sm, 1);
      sm += __shfl_xor(sm, 2);
      sm += __shfl_xor(sm, 4);
      sm += __shfl_xor(sm, 8);
      rSum[r] = rSum[r] * corr + sm;
#pragma unroll
      for (int nf = 0; nf < 4; ++nf) Oa[nf][r] *= corr;
    }

    // P: D-layout f32 -> A-layout f16 via wave-private LDS rows
#pragma unroll
    for (int nf = 0; nf < 4; ++nf)
#pragma unroll
      for (int r = 0; r < 8; ++r)
        sP[(wRow + ((lane >> 4) << 3) + r) * AT_KP + nf * 16 + (lane & 15)] =
            (_Float16)S[nf][r];
    asm volatile("s_wait_dscnt 0" ::: "memory");  // wave-local LDS RAW

    // O += P @ V   (B-frag: n = dh, K-dim = s, contiguous in sV rows)
#pragma unroll
    for (int kk = 0; kk < 2; ++kk) {
      v16h pa = load_a_frag(sP, AT_KP, wRow, kk * 32, lane);
#pragma unroll
      for (int nf = 0; nf < 4; ++nf) {
        v16h bv = load_b_frag(sV, AT_KP, nf * 16, kk * 32, lane);
        Oa[nf] = __builtin_amdgcn_wmma_f32_16x16x32_f16(false, pa, false, bv,
                                                        (short)0, Oa[nf], false, false);
      }
    }
  }

  // normalize and store as f16 [B*T, C] (col = h*64 + dh) for the proj GEMM
  const int b = bh >> 4, h = bh & 15;
#pragma unroll
  for (int r = 0; r < 8; ++r) rSum[r] = 1.f / rSum[r];
#pragma unroll
  for (int nf = 0; nf < 4; ++nf)
#pragma unroll
    for (int r = 0; r < 8; ++r) {
      const int t  = r0 + wRow + ((lane >> 4) << 3) + r;
      const int cc = h * NB_DH + nf * 16 + (lane & 15);
      O[(size_t)(b * NB_T + t) * NB_C + cc] = (_Float16)(Oa[nf][r] * rSum[r]);
    }
}

// ---------------------------------------------------------------- launch
extern "C" void kernel_launch(void* const* d_in, const int* in_sizes, int n_in,
                              void* d_out, int out_size, void* d_ws, size_t ws_size,
                              hipStream_t stream) {
  (void)in_sizes; (void)n_in; (void)out_size; (void)ws_size;
  const float* x   = (const float*)d_in[0];
  const float* Wq  = (const float*)d_in[1];
  const float* Wk  = (const float*)d_in[2];
  const float* Wv  = (const float*)d_in[3];
  const float* Wp  = (const float*)d_in[4];
  const float* bp  = (const float*)d_in[5];
  const float* W1  = (const float*)d_in[6];
  const float* b1  = (const float*)d_in[7];
  const float* W2  = (const float*)d_in[8];
  const float* b2  = (const float*)d_in[9];
  const float* g1  = (const float*)d_in[10];
  const float* be1 = (const float*)d_in[11];
  const float* g2  = (const float*)d_in[12];
  const float* be2 = (const float*)d_in[13];

  const size_t CC = (size_t)NB_C * NB_C;          // 1M
  const size_t C4 = (size_t)NB_C * 4 * NB_C;      // 4M
  const size_t MC = (size_t)NB_M * NB_C;          // 8M
  const size_t MF = (size_t)NB_M * 4 * NB_C;      // 32M

  char* wsp = (char*)d_ws;
  size_t off = 0;
  auto take = [&](size_t bytes) {
    char* p = wsp + off;
    off = (off + bytes + 255) & ~(size_t)255;
    return p;
  };
  // All weight buffers hold W^T in f16: [N][K] row-major.
  _Float16* wq16 = (_Float16*)take(CC * 2);
  _Float16* wk16 = (_Float16*)take(CC * 2);
  _Float16* wv16 = (_Float16*)take(CC * 2);
  _Float16* wp16 = (_Float16*)take(CC * 2);
  _Float16* w116 = (_Float16*)take(C4 * 2);   // [4096][1024]
  _Float16* w216 = (_Float16*)take(C4 * 2);   // [1024][4096]
  _Float16* hA   = (_Float16*)take(MC * 2);   // LN1 output
  _Float16* qb   = (_Float16*)take(MC * 2);   // [B,H,T,DH]
  _Float16* kb   = (_Float16*)take(MC * 2);   // [B,H,T,DH]
  _Float16* vtb  = (_Float16*)take(MC * 2);   // [B,H,DH,T]
  _Float16* atb  = (_Float16*)take(MC * 2);   // attention out [B*T, C]
  _Float16* h2   = (_Float16*)take(MC * 2);   // LN2 output
  _Float16* ffb  = (_Float16*)take(MF * 2);   // relu(h2@W1+b1)
  float*    x2   = (float*)   take(MC * 4);   // x + attn@Wp + bp

  dim3 tCC(NB_C / 32, NB_C / 32);                       // 32 x 32
  cvt_t_kernel<<<tCC, 256, 0, stream>>>(Wq, wq16, NB_C, NB_C);
  cvt_t_kernel<<<tCC, 256, 0, stream>>>(Wk, wk16, NB_C, NB_C);
  cvt_t_kernel<<<tCC, 256, 0, stream>>>(Wv, wv16, NB_C, NB_C);
  cvt_t_kernel<<<tCC, 256, 0, stream>>>(Wp, wp16, NB_C, NB_C);
  cvt_t_kernel<<<dim3(4 * NB_C / 32, NB_C / 32), 256, 0, stream>>>(W1, w116, NB_C, 4 * NB_C);
  cvt_t_kernel<<<dim3(NB_C / 32, 4 * NB_C / 32), 256, 0, stream>>>(W2, w216, 4 * NB_C, NB_C);

  ln_kernel<<<NB_M / 8, 256, 0, stream>>>(x, g1, be1, hA);

  dim3 gCC(NB_C / GM_NB, NB_M / GM_MB);                 // 8 x 64
  gemm_kernel<OUT_QK><<<gCC, 256, 0, stream>>>(hA, wq16, nullptr, nullptr, qb, NB_C, NB_C);
  gemm_kernel<OUT_QK><<<gCC, 256, 0, stream>>>(hA, wk16, nullptr, nullptr, kb, NB_C, NB_C);
  gemm_kernel<OUT_VT><<<gCC, 256, 0, stream>>>(hA, wv16, nullptr, nullptr, vtb, NB_C, NB_C);

  attn_kernel<<<dim3(NB_T / AT_BR, 4 * NB_H), 256, 0, stream>>>(qb, kb, vtb, atb);

  gemm_kernel<OUT_RESID><<<gCC, 256, 0, stream>>>(atb, wp16, bp, x, x2, NB_C, NB_C);

  ln_kernel<<<NB_M / 8, 256, 0, stream>>>(x2, g2, be2, h2);

  dim3 gF1(4 * NB_C / GM_NB, NB_M / GM_MB);             // 32 x 64
  gemm_kernel<OUT_RELU><<<gF1, 256, 0, stream>>>(h2, w116, b1, nullptr, ffb, 4 * NB_C, NB_C);

  gemm_kernel<OUT_RESID><<<gCC, 256, 0, stream>>>(ffb, w216, b2, x2, (float*)d_out,
                                                  NB_C, 4 * NB_C);
}
